// MATE_48284022341992
// MI455X (gfx1250) — matrix operand order repaired and checked by reading, not compiled, not run on hardware
//
#include <hip/hip_runtime.h>
#include <math.h>

// ---------------- problem constants (from reference) ----------------
#define N_NODES 100000
#define DD      172      // D
#define D3      516      // 3*D
#define MSGW    688      // 4*D
#define BB      500      // B
#define KK      10       // K
#define MM      1000     // M (msg_node_ids == arange(M))
#define NB1     1500     // 3*B
#define NB2     15000    // 3*B*K

typedef __attribute__((ext_vector_type(16))) __bf16 v16bf;
typedef __attribute__((ext_vector_type(8)))  float  v8f;

__device__ __forceinline__ __bf16 f2bf(float f) {
  unsigned u = __float_as_uint(f);
  unsigned short h = (unsigned short)((u + 0x7FFFu + ((u >> 16) & 1u)) >> 16); // RNE
  __bf16 r; __builtin_memcpy(&r, &h, sizeof(r));
  return r;
}
__device__ __forceinline__ float sigf(float x) { return 1.0f / (1.0f + expf(-x)); }

// mem-after-update lookup: msg_node_ids == arange(M), so row n<M is h_new[n]
__device__ __forceinline__ float feat0_at(int n, int c,
                                          const float* __restrict__ h_new,
                                          const float* __restrict__ memory,
                                          const float* __restrict__ node_feats) {
  float m = (n < MM) ? h_new[(size_t)n * DD + c] : memory[(size_t)n * DD + c];
  return m + node_feats[(size_t)n * DD + c];
}

// ---------------- WMMA tiled GEMM -----------------------------------
// C[M,N] = act( A[M,K] * B  + bias_scale*bias[N] )  (fp32 in/out, bf16 WMMA, f32 acc)
// B_NK=true : B row-major [N,K]  => C = A @ B^T   (x @ W.T pattern)
// B_NK=false: B row-major [K,N]  => C = A @ B
// Block: 128 threads = 4 waves; tile 64(M) x 64(N); K-step 32.
// LDS tiles are stored FRAGMENT-READY: each (wave|nt, lane) owns 16 contiguous
// bf16 (32 bytes), so a fragment load is one aligned v16bf read (2x ds_load_b128)
// instead of 16 scalar ds_load_u16 + packing movs.
template <bool B_NK>
__global__ __launch_bounds__(128)
void gemm_wmma(const float* __restrict__ A, const float* __restrict__ Bm,
               const float* __restrict__ bias, float* __restrict__ C,
               int Mr, int Nc, int Kd, float bias_scale, int act) {
  __shared__ __align__(32) __bf16 As[4 * 32 * 16];  // [wave][lane][e] : 64x32 A tile
  __shared__ __align__(32) __bf16 Bs[4 * 32 * 16];  // [nt]  [lane][e] : 32x64 B tile

  const int tid  = threadIdx.x;
  const int wave = tid >> 5;
  const int lane = tid & 31;
  const int hsel = lane >> 4;     // wave32 half select
  const int l16  = lane & 15;
  const int mBlk = blockIdx.y * 64;
  const int nBlk = blockIdx.x * 64;

  v8f acc[4] = {};                // 4 N-subtiles of 16 per wave

  for (int k0 = 0; k0 < Kd; k0 += 32) {
    // ---- prefetch next K-tile (global_prefetch_b8) ----
    int nk0 = k0 + 32;
    if (nk0 < Kd) {
      int pr = tid & 63;                       // one row per thread (covers 64 rows)
      int gm = mBlk + pr;
      if (gm < Mr) __builtin_prefetch(&A[(size_t)gm * Kd + nk0], 0, 0);
      if (B_NK) {
        int gn = nBlk + pr;
        if (gn < Nc) __builtin_prefetch(&Bm[(size_t)gn * Kd + nk0], 0, 0);
      } else {
        int pk = tid & 31;
        if (nk0 + pk < Kd) __builtin_prefetch(&Bm[(size_t)(nk0 + pk) * Nc + nBlk], 0, 0);
      }
    }

    // ---- stage A tile: 64 rows x 8 float4-chunks; chunk stays in one e-run ----
    for (int idx = tid; idx < 512; idx += 128) {
      int r  = idx >> 3;            // tile row 0..63
      int kc = (idx & 7) << 2;      // k chunk start (mult of 4)
      int gm = mBlk + r, gk = k0 + kc;
      float x0 = 0.f, x1 = 0.f, x2 = 0.f, x3 = 0.f;
      if (gm < Mr) {
        if (gk + 3 < Kd) {
          float4 v = *(const float4*)(A + (size_t)gm * Kd + gk);
          x0 = v.x; x1 = v.y; x2 = v.z; x3 = v.w;
        } else {
          if (gk     < Kd) x0 = A[(size_t)gm * Kd + gk];
          if (gk + 1 < Kd) x1 = A[(size_t)gm * Kd + gk + 1];
          if (gk + 2 < Kd) x2 = A[(size_t)gm * Kd + gk + 2];
          if (gk + 3 < Kd) x3 = A[(size_t)gm * Kd + gk + 3];
        }
      }
      int wv = r >> 4, m15 = r & 15;
      int hs = kc >> 4;
      int e0 = kc & 15;
      __bf16* d = &As[((wv * 32) + (hs * 16 + m15)) * 16 + e0];
      d[0] = f2bf(x0); d[1] = f2bf(x1); d[2] = f2bf(x2); d[3] = f2bf(x3);
    }

    // ---- stage B tile (K-major fragments: lane=(n&15)+(k>>4)*16, e=k&15) ----
    if (B_NK) {
      // B[n][k]: global-contiguous over k -> float4 over k, contiguous LDS e-run
      for (int idx = tid; idx < 512; idx += 128) {
        int n  = idx >> 3;           // 0..63
        int kc = (idx & 7) << 2;     // 0..28
        int gn = nBlk + n, gk = k0 + kc;
        float x0 = 0.f, x1 = 0.f, x2 = 0.f, x3 = 0.f;
        if (gn < Nc) {
          if (gk + 3 < Kd) {
            float4 v = *(const float4*)(Bm + (size_t)gn * Kd + gk);
            x0 = v.x; x1 = v.y; x2 = v.z; x3 = v.w;
          } else {
            if (gk     < Kd) x0 = Bm[(size_t)gn * Kd + gk];
            if (gk + 1 < Kd) x1 = Bm[(size_t)gn * Kd + gk + 1];
            if (gk + 2 < Kd) x2 = Bm[(size_t)gn * Kd + gk + 2];
            if (gk + 3 < Kd) x3 = Bm[(size_t)gn * Kd + gk + 3];
          }
        }
        int nt = n >> 4;
        int l  = (n & 15) + (((kc >> 4) & 1) << 4);
        int e0 = kc & 15;
        __bf16* d = &Bs[((nt * 32) + l) * 16 + e0];
        d[0] = f2bf(x0); d[1] = f2bf(x1); d[2] = f2bf(x2); d[3] = f2bf(x3);
      }
    } else {
      // B[k][n]: global-contiguous over n -> float4 over n, 4 scattered LDS stores
      for (int idx = tid; idx < 512; idx += 128) {
        int k  = idx >> 4;           // 0..31
        int nc = (idx & 15) << 2;    // 0..60
        int gk = k0 + k, gn = nBlk + nc;
        float x[4] = {0.f, 0.f, 0.f, 0.f};
        if (gk < Kd) {
          if (gn + 3 < Nc) {
            float4 v = *(const float4*)(Bm + (size_t)gk * Nc + gn);
            x[0] = v.x; x[1] = v.y; x[2] = v.z; x[3] = v.w;
          } else {
#pragma unroll
            for (int q = 0; q < 4; ++q)
              if (gn + q < Nc) x[q] = Bm[(size_t)gk * Nc + gn + q];
          }
        }
        int hi = (k >> 4) << 4;      // 0 or 16 (lane offset from k)
        int e  = k & 15;
#pragma unroll
        for (int q = 0; q < 4; ++q) {
          int n  = nc + q;
          int nt = n >> 4;
          int l  = (n & 15) + hi;
          Bs[((nt * 32) + l) * 16 + e] = f2bf(x[q]);
        }
      }
    }
    __syncthreads();

    // ---- fragments: single aligned 32B LDS reads ----
    const v16bf afrag = *reinterpret_cast<const v16bf*>(&As[(wave * 32 + lane) * 16]);
#pragma unroll
    for (int nt = 0; nt < 4; ++nt) {
      const v16bf bfrag = *reinterpret_cast<const v16bf*>(&Bs[(nt * 32 + lane) * 16]);
      acc[nt] = __builtin_amdgcn_wmma_f32_16x16x32_bf16(
          false, afrag, false, bfrag, (short)0, acc[nt], false, false);
    }
    __syncthreads();
  }

  // epilogue: C layout — lane half selects M 0-7 / 8-15, N = l16
#pragma unroll
  for (int nt = 0; nt < 4; ++nt) {
    int gn = nBlk + nt * 16 + l16;
    if (gn >= Nc) continue;
    float bv = bias ? bias_scale * bias[gn] : 0.0f;
#pragma unroll
    for (int r = 0; r < 8; ++r) {
      int gm = mBlk + wave * 16 + hsel * 8 + r;
      if (gm >= Mr) continue;
      float v = acc[nt][r] + bv;
      if (act == 1) v = fmaxf(v, 0.0f);
      C[(size_t)gm * Nc + gn] = v;
    }
  }
}

// ---------------- GRU gate combine ----------------------------------
__global__ void gru_combine(const float* __restrict__ gi, const float* __restrict__ gh,
                            const float* __restrict__ memory, float* __restrict__ h_new) {
  int idx = blockIdx.x * blockDim.x + threadIdx.x;
  if (idx >= MM * DD) return;
  int m = idx / DD, j = idx % DD;
  const float* gim = gi + (size_t)m * D3;
  const float* ghm = gh + (size_t)m * D3;
  float r = sigf(gim[j] + ghm[j]);
  float z = sigf(gim[DD + j] + ghm[DD + j]);
  float g = tanhf(gim[2 * DD + j] + r * ghm[2 * DD + j]);
  float h = memory[(size_t)m * DD + j];
  h_new[idx] = (1.0f - z) * g + z * h;
}

// ---------------- neighbor pooling (the linear-sum hoist) ------------
__global__ void pool_neighbors(int nrows, int mode,
                               const int* __restrict__ nbr_ids,
                               const int* __restrict__ nbr_eidx,
                               const float* __restrict__ nbr_t,
                               const float* __restrict__ edge_times, int tsdiv,
                               const float* __restrict__ emb_rows,
                               const float* __restrict__ h_new,
                               const float* __restrict__ memory,
                               const float* __restrict__ node_feats,
                               const float* __restrict__ edge_feats,
                               const float* __restrict__ tw, const float* __restrict__ tb,
                               float* __restrict__ pooled) {
  int i = blockIdx.x;
  if (i >= nrows) return;
  __shared__ int   s_id[KK];
  __shared__ int   s_ei[KK];
  __shared__ float s_dt[KK];
  float ts = edge_times[(i / tsdiv) % BB];
  if (threadIdx.x < KK) {
    int j = threadIdx.x;
    if (mode == 0) s_id[j] = nbr_ids[(size_t)i * KK + j];
    s_ei[j] = nbr_eidx[(size_t)i * KK + j];
    s_dt[j] = ts - nbr_t[(size_t)i * KK + j];
  }
  __syncthreads();
  for (int c = threadIdx.x; c < D3; c += blockDim.x) {
    float acc = 0.0f;
    if (c < DD) {
      if (mode == 0) {
#pragma unroll
        for (int j = 0; j < KK; ++j)
          acc += feat0_at(s_id[j], c, h_new, memory, node_feats);
      } else {
#pragma unroll
        for (int j = 0; j < KK; ++j)
          acc += emb_rows[((size_t)i * KK + j) * DD + c];
      }
    } else if (c < 2 * DD) {
      float w = tw[c - DD], b = tb[c - DD];
#pragma unroll
      for (int j = 0; j < KK; ++j) acc += cosf(s_dt[j] * w + b);
    } else {
      int cc = c - 2 * DD;
#pragma unroll
      for (int j = 0; j < KK; ++j) acc += edge_feats[(size_t)s_ei[j] * DD + cc];
    }
    pooled[(size_t)i * D3 + c] = acc;
  }
}

// ---------------- s = [h | feat0(src) | cos(time_b)] -----------------
__global__ void build_s(int nrows, const float* __restrict__ hmat,
                        int src_mode, const int* __restrict__ n1,
                        const int* __restrict__ src, const int* __restrict__ dst,
                        const int* __restrict__ neg,
                        const float* __restrict__ h_new, const float* __restrict__ memory,
                        const float* __restrict__ node_feats, const float* __restrict__ tb,
                        float* __restrict__ S) {
  size_t idx = (size_t)blockIdx.x * blockDim.x + threadIdx.x;
  size_t tot = (size_t)nrows * D3;
  if (idx >= tot) return;
  int i = (int)(idx / D3), c = (int)(idx % D3);
  float v;
  if (c < DD) {
    v = hmat[(size_t)i * DD + c];
  } else if (c < 2 * DD) {
    int cc = c - DD;
    int n = (src_mode == 0) ? n1[i]
            : (i < BB ? src[i] : (i < 2 * BB ? dst[i - BB] : neg[i - 2 * BB]));
    v = feat0_at(n, cc, h_new, memory, node_feats);
  } else {
    v = cosf(tb[c - 2 * DD]);        // _time_enc(0) = cos(b)
  }
  S[idx] = v;
}

// ---------------- affinity input: [s_emb|d_emb] ; [s_emb|g_emb] ------
__global__ void build_aff(const float* __restrict__ emb2, float* __restrict__ A) {
  int idx = blockIdx.x * blockDim.x + threadIdx.x;
  if (idx >= 2 * BB * 2 * DD) return;
  int i = idx / (2 * DD), c = idx % (2 * DD);
  int half = (c >= DD) ? 1 : 0;
  int cc = c - half * DD;
  int row;
  if (i < BB) row = half ? (BB + i) : i;                   // pos: (s, d)
  else        row = half ? (2 * BB + (i - BB)) : (i - BB); // neg: (s, g)
  A[idx] = emb2[(size_t)row * DD + cc];
}

// ---------------- sigmoid(h . w2 + b2) -------------------------------
__global__ void aff_reduce(const float* __restrict__ affH, const float* __restrict__ w2,
                           const float* __restrict__ b2, float* __restrict__ out) {
  int i = blockIdx.x * blockDim.x + threadIdx.x;
  if (i >= 2 * BB) return;
  float s = b2[0];
  for (int c = 0; c < DD; ++c) s += affH[(size_t)i * DD + c] * w2[c];
  out[i] = sigf(s);
}

// ---------------- raw = [node_feats[src]; node_feats[dst]] -----------
__global__ void raw_gather(const float* __restrict__ nf, const int* __restrict__ src,
                           const int* __restrict__ dst, float* __restrict__ out) {
  int idx = blockIdx.x * blockDim.x + threadIdx.x;
  if (idx >= 2 * BB * DD) return;
  int i = idx / DD, c = idx % DD;
  int n = (i < BB) ? src[i] : dst[i - BB];
  out[idx] = nf[(size_t)n * DD + c];
}

// ---------------- launcher ------------------------------------------
extern "C" void kernel_launch(void* const* d_in, const int* in_sizes, int n_in,
                              void* d_out, int out_size, void* d_ws, size_t ws_size,
                              hipStream_t stream) {
  (void)in_sizes; (void)n_in; (void)out_size; (void)ws_size;

  const float* node_feats = (const float*)d_in[0];
  const float* edge_feats = (const float*)d_in[1];
  const float* memory_    = (const float*)d_in[2];
  const float* time_w     = (const float*)d_in[3];
  const float* time_b     = (const float*)d_in[4];
  const float* W_ih       = (const float*)d_in[5];
  const float* W_hh       = (const float*)d_in[6];
  const float* b_ih       = (const float*)d_in[7];
  const float* b_hh       = (const float*)d_in[8];
  const float* lin1_w     = (const float*)d_in[9];
  const float* lin1_b     = (const float*)d_in[10];
  const float* lin2_w     = (const float*)d_in[11];
  const float* lin2_b     = (const float*)d_in[12];
  const float* aff_w1     = (const float*)d_in[13];
  const float* aff_b1     = (const float*)d_in[14];
  const float* aff_w2     = (const float*)d_in[15];
  const float* aff_b2     = (const float*)d_in[16];
  const float* dec_w      = (const float*)d_in[17];
  const float* dec_b      = (const float*)d_in[18];
  const float* edge_times = (const float*)d_in[19];
  const float* nbr1_t     = (const float*)d_in[20];
  const float* nbr2_t     = (const float*)d_in[21];
  const float* msgs       = (const float*)d_in[22];
  const int*   src        = (const int*)d_in[24];
  const int*   dst        = (const int*)d_in[25];
  const int*   neg        = (const int*)d_in[26];
  const int*   nbr1_ids   = (const int*)d_in[28];
  const int*   nbr1_eidx  = (const int*)d_in[29];
  const int*   nbr2_ids   = (const int*)d_in[30];
  const int*   nbr2_eidx  = (const int*)d_in[31];

  // workspace carve (~64 MB of f32)
  float* ws   = (float*)d_ws;
  float* gi   = ws; ws += (size_t)MM * D3;
  float* gh   = ws; ws += (size_t)MM * D3;
  float* hnew = ws; ws += (size_t)MM * DD;
  float* big0 = ws; ws += (size_t)NB2 * D3;   // pooled0, later reused as s0
  float* hm0  = ws; ws += (size_t)NB2 * DD;
  float* emb1 = ws; ws += (size_t)NB2 * DD;
  float* big1 = ws; ws += (size_t)NB1 * D3;   // pooled1, later reused as s1
  float* hm1  = ws; ws += (size_t)NB1 * DD;
  float* emb2 = ws; ws += (size_t)NB1 * DD;
  float* affA = ws; ws += (size_t)2 * BB * 2 * DD;
  float* affH = ws; ws += (size_t)2 * BB * DD;

  float* out = (float*)d_out;   // [pos(500) | neg(500) | decode(1000*172) | raw(1000*172)]

  const dim3 blk(128);
  auto gemm_grid = [](int M, int N) { return dim3((N + 63) / 64, (M + 63) / 64); };

  // --- GRU memory update ---
  gemm_wmma<true ><<<gemm_grid(MM, D3), blk, 0, stream>>>(msgs,    W_ih, b_ih, gi, MM, D3, MSGW, 1.0f, 0);
  gemm_wmma<true ><<<gemm_grid(MM, D3), blk, 0, stream>>>(memory_, W_hh, b_hh, gh, MM, D3, DD,   1.0f, 0);
  gru_combine<<<(MM * DD + 255) / 256, 256, 0, stream>>>(gi, gh, memory_, hnew);

  // --- level 0: 15000 targets, K=10 second-hop neighbors ---
  pool_neighbors<<<NB2, 256, 0, stream>>>(NB2, 0, nbr2_ids, nbr2_eidx, nbr2_t,
      edge_times, KK, nullptr, hnew, memory_, node_feats, edge_feats, time_w, time_b, big0);
  gemm_wmma<false><<<gemm_grid(NB2, DD), blk, 0, stream>>>(big0, lin1_w, lin1_b, hm0,
      NB2, DD, D3, (float)KK, 1);                                   // relu, bias*K (sum hoist)
  build_s<<<(int)(((size_t)NB2 * D3 + 255) / 256), 256, 0, stream>>>(NB2, hm0, 0,
      nbr1_ids, nullptr, nullptr, nullptr, hnew, memory_, node_feats, time_b, big0);
  gemm_wmma<false><<<gemm_grid(NB2, DD), blk, 0, stream>>>(big0, lin2_w, lin2_b, emb1,
      NB2, DD, D3, 1.0f, 0);

  // --- level 1: 1500 targets, K=10 first-hop neighbors (emb1 as nbr emb) ---
  pool_neighbors<<<NB1, 256, 0, stream>>>(NB1, 1, nullptr, nbr1_eidx, nbr1_t,
      edge_times, 1, emb1, hnew, memory_, node_feats, edge_feats, time_w, time_b, big1);
  gemm_wmma<false><<<gemm_grid(NB1, DD), blk, 0, stream>>>(big1, lin1_w + (size_t)D3 * DD,
      lin1_b + DD, hm1, NB1, DD, D3, (float)KK, 1);
  build_s<<<(int)(((size_t)NB1 * D3 + 255) / 256), 256, 0, stream>>>(NB1, hm1, 1,
      nullptr, src, dst, neg, hnew, memory_, node_feats, time_b, big1);
  gemm_wmma<false><<<gemm_grid(NB1, DD), blk, 0, stream>>>(big1, lin2_w + (size_t)D3 * DD,
      lin2_b + DD, emb2, NB1, DD, D3, 1.0f, 0);

  // --- affinity MLP -> pos/neg probs ---
  build_aff<<<(2 * BB * 2 * DD + 255) / 256, 256, 0, stream>>>(emb2, affA);
  gemm_wmma<false><<<gemm_grid(2 * BB, DD), blk, 0, stream>>>(affA, aff_w1, aff_b1, affH,
      2 * BB, DD, 2 * DD, 1.0f, 1);
  aff_reduce<<<(2 * BB + 255) / 256, 256, 0, stream>>>(affH, aff_w2, aff_b2, out);

  // --- decode: emb2 rows [0,1000) are exactly [s_emb; d_emb] ---
  gemm_wmma<false><<<gemm_grid(2 * BB, DD), blk, 0, stream>>>(emb2, dec_w, dec_b,
      out + 2 * BB, 2 * BB, DD, DD, 1.0f, 0);

  // --- raw feature gather ---
  raw_gather<<<(2 * BB * DD + 255) / 256, 256, 0, stream>>>(node_feats, src, dst,
      out + 2 * BB + (size_t)2 * BB * DD);
}
